// Model_65085934404102
// MI455X (gfx1250) — compile-verified, matrix operand order checked by reading, not run on hardware
//
#include <hip/hip_runtime.h>
#include <hip/hip_bf16.h>

// ---------------------------------------------------------------------------
// CDNA5 (gfx1250, wave32) WMMA types
// ---------------------------------------------------------------------------
typedef __attribute__((ext_vector_type(16))) __bf16 v16bf;
typedef __attribute__((ext_vector_type(8)))  float  v8f;

#define BM 128
#define BN 64
#define TK 32
#define APAD 8   // row padding (ushorts): row stride 80B keeps 16B alignment

#if defined(__gfx1250__) && __has_builtin(__builtin_amdgcn_global_load_async_to_lds_b128)
#define USE_ASYNC 1
typedef int v4i_vs __attribute__((vector_size(16)));
typedef __attribute__((address_space(1))) v4i_vs GVec;   // global int4
typedef __attribute__((address_space(3))) v4i_vs LVec;   // LDS int4
#else
#define USE_ASYNC 0
#endif

__device__ __forceinline__ unsigned short f2bf(float x) {
    // round-to-nearest-even float -> bf16
    unsigned u = __float_as_uint(x);
    unsigned r = (u + 0x7FFFu + ((u >> 16) & 1u)) >> 16;
    return (unsigned short)r;
}
__device__ __forceinline__ unsigned pack2bf(float a, float b) {
    return (unsigned)f2bf(a) | ((unsigned)f2bf(b) << 16);
}

__device__ __forceinline__ v16bf load_frag(const unsigned short* p0,
                                           const unsigned short* p1) {
    union { uint4 q[2]; v16bf v; } t;
    t.q[0] = *(const uint4*)p0;   // ds_load_b128
    t.q[1] = *(const uint4*)p1;   // ds_load_b128
    return t.v;
}

// Ordered-uint encoding so unsigned atomicMax implements float max.
__device__ __forceinline__ unsigned fkey(float f) {
    unsigned u = __float_as_uint(f);
    return (u & 0x80000000u) ? ~u : (u | 0x80000000u);
}
__device__ __forceinline__ float fdec(unsigned k) {
    unsigned u = (k & 0x80000000u) ? (k & 0x7FFFFFFFu) : ~k;
    return __uint_as_float(u);
}

// ---------------------------------------------------------------------------
// fp32 -> bf16 conversion (optional fused ReLU). n multiple of 8.
// ---------------------------------------------------------------------------
__global__ void cvt_bf16_kernel(const float* __restrict__ x,
                                unsigned short* __restrict__ y,
                                size_t n, int relu)
{
    size_t i = ((size_t)blockIdx.x * 256 + threadIdx.x) * 8;
    if (i >= n) return;
    float4 a = *(const float4*)(x + i);
    float4 b = *(const float4*)(x + i + 4);
    if (relu) {
        a.x = fmaxf(a.x, 0.f); a.y = fmaxf(a.y, 0.f);
        a.z = fmaxf(a.z, 0.f); a.w = fmaxf(a.w, 0.f);
        b.x = fmaxf(b.x, 0.f); b.y = fmaxf(b.y, 0.f);
        b.z = fmaxf(b.z, 0.f); b.w = fmaxf(b.w, 0.f);
    }
    uint4 p;
    p.x = pack2bf(a.x, a.y); p.y = pack2bf(a.z, a.w);
    p.z = pack2bf(b.x, b.y); p.w = pack2bf(b.z, b.w);
    *(uint4*)(y + i) = p;
}

// Weight transpose + convert: W[K,N] fp32 -> Wt[N,K] bf16 (weights are tiny).
__global__ void wtrans_kernel(const float* __restrict__ w,
                              unsigned short* __restrict__ wt, int K, int N)
{
    int idx = blockIdx.x * 256 + threadIdx.x;
    if (idx >= K * N) return;
    int n = idx / K, k = idx - n * K;
    wt[idx] = f2bf(w[(size_t)k * N + n]);
}

// ---------------------------------------------------------------------------
// WMMA GEMM: C[M,N] = A_bf16[M,K] @ Wt_bf16[N,K]^T (+bias) (+relu)
// Block tile 128x64, 8 wave32 in 4x2 grid, each wave 32x32 = 4 accumulators
// -> 4 x v_wmma_f32_16x16x32_bf16 per K-step. K mult of 32, N mult of 64.
// Double-buffered LDS; tiles streamed with GLOBAL_LOAD_ASYNC_TO_LDS_B128
// (ASYNCcnt) when available, else global b128 + ds b128. Out-of-range rows
// are clamped to M-1 (finite data, discarded at the epilogue) so staging is
// branch-free. Output: fp32 (Cf) or bf16 (Cb) selected by non-null pointer.
// ---------------------------------------------------------------------------
__global__ __launch_bounds__(256)
void gemm_wmma_kernel(const unsigned short* __restrict__ A,
                      const unsigned short* __restrict__ Wt,
                      const float* __restrict__ bias,
                      float* __restrict__ Cf, unsigned short* __restrict__ Cb,
                      int M, int K, int N, int relu)
{
    __shared__ unsigned short As[2][BM][TK + APAD];
    __shared__ unsigned short Bs[2][BN][TK + APAD];

    const int tid  = threadIdx.x;
    const int lane = tid & 31;
    const int wave = tid >> 5;
    const int wr   = wave >> 1;   // 0..3 : 32-row band
    const int wc   = wave & 1;    // 0..1 : 32-col band
    const int bm   = blockIdx.x * BM;
    const int bn   = blockIdx.y * BN;

    v8f acc[2][2] = {};

    const int arow = tid >> 1;          // 0..127
    const int aseg = (tid & 1) * 16;    // 0 or 16 (bf16 elements)
    const int brow = tid >> 2;          // 0..63
    const int bseg = (tid & 3) * 8;     // 0,8,16,24

    const int kh  = lane >> 4;
    const int l16 = lane & 15;

    // clamp rows beyond M to the last valid row (results discarded later)
    int gr = bm + arow; if (gr >= M) gr = M - 1;
    const unsigned short* arowp = A  + (size_t)gr * K;
    const unsigned short* browp = Wt + (size_t)(bn + brow) * K;

    auto stage = [&](int p, int k0) {
        const unsigned short* ap = arowp + k0 + aseg;
        const unsigned short* wp = browp + k0 + bseg;
#if USE_ASYNC
        __builtin_amdgcn_global_load_async_to_lds_b128(
            (GVec*)(unsigned long long)(const void*)ap,
            (LVec*)(unsigned)(unsigned long long)(void*)&As[p][arow][aseg], 0, 0);
        __builtin_amdgcn_global_load_async_to_lds_b128(
            (GVec*)(unsigned long long)(const void*)(ap + 8),
            (LVec*)(unsigned)(unsigned long long)(void*)&As[p][arow][aseg + 8], 0, 0);
        __builtin_amdgcn_global_load_async_to_lds_b128(
            (GVec*)(unsigned long long)(const void*)wp,
            (LVec*)(unsigned)(unsigned long long)(void*)&Bs[p][brow][bseg], 0, 0);
#else
        uint4 v0 = *(const uint4*)ap;        // global_load_b128
        uint4 v1 = *(const uint4*)(ap + 8);
        uint4 w0 = *(const uint4*)wp;
        *(uint4*)&As[p][arow][aseg]     = v0; // ds_store_b128
        *(uint4*)&As[p][arow][aseg + 8] = v1;
        *(uint4*)&Bs[p][brow][bseg]     = w0;
#endif
    };

    stage(0, 0);
    for (int k0 = 0; k0 < K; k0 += TK) {
        const int  cur  = (k0 / TK) & 1;
        const bool more = (k0 + TK) < K;
        if (more) stage(cur ^ 1, k0 + TK);   // prefetch next tile (other buffer)
#if USE_ASYNC
        // retire current tile's 3 async transfers (next tile's 3 stay in flight)
        if (more) asm volatile("s_wait_asynccnt 0x3" ::: "memory");
        else      asm volatile("s_wait_asynccnt 0x0" ::: "memory");
#endif
        __syncthreads();                      // current tile visible block-wide

        v16bf a0 = load_frag(&As[cur][wr * 32 + l16][kh * 8],
                             &As[cur][wr * 32 + l16][16 + kh * 8]);
        v16bf a1 = load_frag(&As[cur][wr * 32 + 16 + l16][kh * 8],
                             &As[cur][wr * 32 + 16 + l16][16 + kh * 8]);
        v16bf b0 = load_frag(&Bs[cur][wc * 32 + l16][kh * 8],
                             &Bs[cur][wc * 32 + l16][16 + kh * 8]);
        v16bf b1 = load_frag(&Bs[cur][wc * 32 + 16 + l16][kh * 8],
                             &Bs[cur][wc * 32 + 16 + l16][16 + kh * 8]);

        acc[0][0] = __builtin_amdgcn_wmma_f32_16x16x32_bf16(
                        false, a0, false, b0, (short)0, acc[0][0], false, false);
        acc[0][1] = __builtin_amdgcn_wmma_f32_16x16x32_bf16(
                        false, a0, false, b1, (short)0, acc[0][1], false, false);
        acc[1][0] = __builtin_amdgcn_wmma_f32_16x16x32_bf16(
                        false, a1, false, b0, (short)0, acc[1][0], false, false);
        acc[1][1] = __builtin_amdgcn_wmma_f32_16x16x32_bf16(
                        false, a1, false, b1, (short)0, acc[1][1], false, false);
        __syncthreads();   // all reads of 'cur' done before it is overwritten
    }

    // ---- epilogue: C layout lanes 0-15 M=j, lanes 16-31 M=j+8 ----
    #pragma unroll
    for (int s = 0; s < 2; ++s) {
        #pragma unroll
        for (int t = 0; t < 2; ++t) {
            #pragma unroll
            for (int j = 0; j < 8; ++j) {
                const int r = bm + wr * 32 + s * 16 + j + kh * 8;
                if (r < M) {
                    const int c = bn + wc * 32 + t * 16 + l16;
                    float v = acc[s][t][j];
                    if (bias) v += bias[c];
                    if (relu) v = fmaxf(v, 0.0f);
                    if (Cb) Cb[(size_t)r * N + c] = f2bf(v);
                    else    Cf[(size_t)r * N + c] = v;
                }
            }
        }
    }
}

// ---------------------------------------------------------------------------
// Attention stage kernels (one wave32 per edge; lane covers 4 of 128 channels)
// ---------------------------------------------------------------------------
__global__ void init_mden_kernel(unsigned* mkey, float* den, int n)
{
    int i = blockIdx.x * 256 + threadIdx.x;
    if (i < n) { mkey[i] = 0u; den[i] = 0.0f; }
}

__global__ __launch_bounds__(256)
void attn_logits_kernel(const float* __restrict__ q, const float* __restrict__ kl,
                        const float* __restrict__ ep,
                        const int* __restrict__ src, const int* __restrict__ dst,
                        float* __restrict__ logits, unsigned* __restrict__ mkey, int E)
{
    int e = blockIdx.x * 8 + (threadIdx.x >> 5);
    if (e >= E) return;
    int lane = threadIdx.x & 31;
    int s = src[e], d = dst[e];
    float4 q4 = ((const float4*)(q  + (size_t)d * 128))[lane];
    float4 k4 = ((const float4*)(kl + (size_t)s * 128))[lane];
    float4 e4 = ((const float4*)(ep + (size_t)e * 128))[lane];
    float p = q4.x * (k4.x + e4.x) + q4.y * (k4.y + e4.y)
            + q4.z * (k4.z + e4.z) + q4.w * (k4.w + e4.w);
    #pragma unroll
    for (int off = 16; off > 0; off >>= 1) p += __shfl_xor(p, off, 32);
    if (lane == 0) {
        float l = p * 0.08838834764831845f;   // 1/sqrt(128)
        logits[e] = l;
        atomicMax(&mkey[d], fkey(l));
    }
}

__global__ void attn_exp_kernel(float* __restrict__ logits,
                                const int* __restrict__ dst,
                                const unsigned* __restrict__ mkey,
                                float* __restrict__ den, int E)
{
    int e = blockIdx.x * 256 + threadIdx.x;
    if (e >= E) return;
    int d = dst[e];
    float x = expf(logits[e] - fdec(mkey[d]));
    logits[e] = x;                 // logits buffer becomes ex
    atomicAdd(&den[d], x);
}

__global__ __launch_bounds__(256)
void attn_agg_kernel(const float* __restrict__ ex, const float* __restrict__ den,
                     const float* __restrict__ vl, const float* __restrict__ ep,
                     const int* __restrict__ src, const int* __restrict__ dst,
                     float* __restrict__ out, int E)
{
    int e = blockIdx.x * 8 + (threadIdx.x >> 5);
    if (e >= E) return;
    int lane = threadIdx.x & 31;
    int s = src[e], d = dst[e];
    float alpha = ex[e] / (den[d] + 1e-16f);
    float4 v4 = ((const float4*)(vl + (size_t)s * 128))[lane];
    float4 e4 = ((const float4*)(ep + (size_t)e * 128))[lane];
    float* o = out + (size_t)d * 128 + lane * 4;
    atomicAdd(o + 0, alpha * (v4.x + e4.x));
    atomicAdd(o + 1, alpha * (v4.y + e4.y));
    atomicAdd(o + 2, alpha * (v4.z + e4.z));
    atomicAdd(o + 3, alpha * (v4.w + e4.w));
}

// ---------------------------------------------------------------------------
// Decoder helpers
// ---------------------------------------------------------------------------
__global__ void gather_z_kernel(const float* __restrict__ zu, const float* __restrict__ zb,
                                const int* __restrict__ row, const int* __restrict__ col,
                                unsigned short* __restrict__ z, int L)
{
    size_t idx = (size_t)blockIdx.x * 256 + threadIdx.x;
    size_t total = (size_t)L * 64;          // 64 float4 per 256-wide row
    if (idx >= total) return;
    int l = (int)(idx >> 6);
    int j = (int)(idx & 63);
    float4 v;
    if (j < 32) v = ((const float4*)(zu + (size_t)row[l] * 128))[j];
    else        v = ((const float4*)(zb + (size_t)col[l] * 128))[j - 32];
    uint2 p;
    p.x = pack2bf(v.x, v.y);
    p.y = pack2bf(v.z, v.w);
    *(uint2*)(z + (size_t)l * 256 + (size_t)j * 4) = p;
}

__global__ void lin4_kernel(const float* __restrict__ h, const float* __restrict__ w,
                            const float* __restrict__ b, float* __restrict__ out, int L)
{
    int idx = blockIdx.x * 256 + threadIdx.x;
    int l = idx / 5, c = idx - l * 5;
    if (l >= L) return;
    const float* hr = h + (size_t)l * 64;
    float acc = b[c];
    #pragma unroll
    for (int k = 0; k < 64; ++k) acc += hr[k] * w[k * 5 + c];
    out[(size_t)l * 5 + c] = acc;
}

// ---------------------------------------------------------------------------
// Host side
// ---------------------------------------------------------------------------
extern "C" void kernel_launch(void* const* d_in, const int* in_sizes, int n_in,
                              void* d_out, int out_size, void* d_ws, size_t ws_size,
                              hipStream_t stream)
{
    (void)n_in; (void)out_size; (void)ws_size;
    typedef unsigned short ushort_t;
    const int Hh = 128;
    const float* user_emb  = (const float*)d_in[0];
    const float* book_emb  = (const float*)d_in[1];
    const float* edge_attr = (const float*)d_in[2];
    const int NU = in_sizes[0] / Hh;
    const int NB = in_sizes[1] / Hh;
    const int E  = in_sizes[2] / Hh;

    // params flattened leaf-wise in sorted-key (pytree) order:
    // conv1_bu, conv1_ub, conv2_bu, conv2_ub, dec
    // per conv (sorted): e.w, k.b, k.w, q.b, q.w, skip.b, skip.w, v.b, v.w
    struct ConvW { const float *ew,*kb,*kw,*qb,*qw,*sb,*sw,*vb,*vw; };
    auto loadConv = [&](int base) -> ConvW {
        return ConvW{ (const float*)d_in[base+0], (const float*)d_in[base+1],
                      (const float*)d_in[base+2], (const float*)d_in[base+3],
                      (const float*)d_in[base+4], (const float*)d_in[base+5],
                      (const float*)d_in[base+6], (const float*)d_in[base+7],
                      (const float*)d_in[base+8] };
    };
    ConvW c1bu = loadConv(3);
    ConvW c1ub = loadConv(12);
    ConvW c2bu = loadConv(21);
    ConvW c2ub = loadConv(30);
    const float* l1b = (const float*)d_in[39]; const float* l1w = (const float*)d_in[40];
    const float* l2b = (const float*)d_in[41]; const float* l2w = (const float*)d_in[42];
    const float* l3b = (const float*)d_in[43]; const float* l3w = (const float*)d_in[44];
    const float* l4b = (const float*)d_in[45]; const float* l4w = (const float*)d_in[46];
    const int* edge_index  = (const int*)d_in[47];
    const int* label_index = (const int*)d_in[48];
    const int* srcI = edge_index;          // user idx
    const int* dstI = edge_index + E;      // book idx
    const int L = in_sizes[48] / 2;
    const int* rowI = label_index;
    const int* colI = label_index + L;

    // ---- workspace carve ----
    char* wsb = (char*)d_ws;
    size_t off = 0;
    auto carve = [&](size_t bytes) -> char* {
        char* p = wsb + off;
        off += (bytes + 255) & ~(size_t)255;
        return p;
    };
    float*    eproj  = (float*)carve((size_t)E * 128 * 4);   // reused by decoder
    ushort_t* ea_bf  = (ushort_t*)carve((size_t)E  * 128 * 2);
    ushort_t* xu_bf  = (ushort_t*)carve((size_t)NU * 128 * 2);
    ushort_t* xb_bf  = (ushort_t*)carve((size_t)NB * 128 * 2);
    float*    qbuf   = (float*)carve((size_t)NU * 128 * 4);
    float*    klbuf  = (float*)carve((size_t)NU * 128 * 4);
    float*    vlbuf  = (float*)carve((size_t)NU * 128 * 4);
    float*    u1     = (float*)carve((size_t)NU * 128 * 4);
    float*    b1     = (float*)carve((size_t)NB * 128 * 4);
    float*    u2     = (float*)carve((size_t)NU * 128 * 4);
    float*    b2     = (float*)carve((size_t)NB * 128 * 4);
    float*    logits = (float*)carve((size_t)E * 4);
    unsigned* mkey   = (unsigned*)carve((size_t)NU * 4);
    float*    den    = (float*)carve((size_t)NU * 4);
    ushort_t* wt0    = (ushort_t*)carve(32768 * 2);
    ushort_t* wt1    = (ushort_t*)carve(32768 * 2);
    ushort_t* wt2    = (ushort_t*)carve(32768 * 2);
    ushort_t* wt3    = (ushort_t*)carve(32768 * 2);
    ushort_t* wt4    = (ushort_t*)carve(32768 * 2);
    // decoder overlay into e-proj region: z(bf16)+h1(bf16)+h2(bf16)+h3(f32)
    ushort_t* z_bf  = (ushort_t*)eproj;
    ushort_t* h1_bf = z_bf + (size_t)L * 256;
    ushort_t* h2_bf = h1_bf + (size_t)L * 128;
    float*    h3    = (float*)(h2_bf + (size_t)L * 256);

    auto cvt = [&](const float* x, ushort_t* y, size_t n, int relu) {
        cvt_bf16_kernel<<<(unsigned)((n / 8 + 255) / 256), 256, 0, stream>>>(x, y, n, relu);
    };
    auto wtrans = [&](const float* w, ushort_t* wt, int K, int N) {
        wtrans_kernel<<<(K * N + 255) / 256, 256, 0, stream>>>(w, wt, K, N);
    };
    auto gemm = [&](const ushort_t* A, const ushort_t* Wt, const float* bias,
                    float* Cf, ushort_t* Cb, int M, int K, int N, int relu) {
        dim3 g((M + BM - 1) / BM, N / BN);
        gemm_wmma_kernel<<<g, 256, 0, stream>>>(A, Wt, bias, Cf, Cb, M, K, N, relu);
    };

    auto conv = [&](const ConvW& p, const ushort_t* xs, int Ns, const ushort_t* xd, int Nd,
                    const int* sA, const int* dA, float* out) {
        wtrans(p.ew, wt0, 128, 128);
        wtrans(p.qw, wt1, 128, 128);
        wtrans(p.kw, wt2, 128, 128);
        wtrans(p.vw, wt3, 128, 128);
        wtrans(p.sw, wt4, 128, 128);
        gemm(ea_bf, wt0, nullptr, eproj, nullptr, E,  128, 128, 0);
        gemm(xd,    wt1, p.qb,    qbuf,  nullptr, Nd, 128, 128, 0);
        gemm(xs,    wt2, p.kb,    klbuf, nullptr, Ns, 128, 128, 0);
        gemm(xs,    wt3, p.vb,    vlbuf, nullptr, Ns, 128, 128, 0);
        gemm(xd,    wt4, p.sb,    out,   nullptr, Nd, 128, 128, 0);  // out = skip
        init_mden_kernel<<<(Nd + 255) / 256, 256, 0, stream>>>(mkey, den, Nd);
        attn_logits_kernel<<<(E + 7) / 8, 256, 0, stream>>>(qbuf, klbuf, eproj,
                                                            sA, dA, logits, mkey, E);
        attn_exp_kernel<<<(E + 255) / 256, 256, 0, stream>>>(logits, dA, mkey, den, E);
        attn_agg_kernel<<<(E + 7) / 8, 256, 0, stream>>>(logits, den, vlbuf, eproj,
                                                         sA, dA, out, E);
    };

    // one-time bf16 conversions
    cvt(edge_attr, ea_bf, (size_t)E * 128, 0);
    cvt(user_emb,  xu_bf, (size_t)NU * 128, 0);
    cvt(book_emb,  xb_bf, (size_t)NB * 128, 0);

    // layer 1
    conv(c1ub, xu_bf, NU, xb_bf, NB, srcI, dstI, b1);
    conv(c1bu, xb_bf, NB, xu_bf, NU, dstI, srcI, u1);
    // relu + convert layer-1 outputs for layer-2 inputs
    cvt(u1, xu_bf, (size_t)NU * 128, 1);
    cvt(b1, xb_bf, (size_t)NB * 128, 1);
    // layer 2
    conv(c2ub, xu_bf, NU, xb_bf, NB, srcI, dstI, b2);
    conv(c2bu, xb_bf, NB, xu_bf, NU, dstI, srcI, u2);

    // decoder
    {
        size_t total = (size_t)L * 64;
        gather_z_kernel<<<(unsigned)((total + 255) / 256), 256, 0, stream>>>(
            u2, b2, rowI, colI, z_bf, L);
    }
    wtrans(l1w, wt0, 256, 128);
    wtrans(l2w, wt1, 128, 256);
    wtrans(l3w, wt2, 256, 64);
    gemm(z_bf,  wt0, l1b, nullptr, h1_bf, L, 256, 128, 1);
    gemm(h1_bf, wt1, l2b, nullptr, h2_bf, L, 128, 256, 1);
    gemm(h2_bf, wt2, l3b, h3,      nullptr, L, 256, 64, 1);
    {
        int tot = L * 5;
        lin4_kernel<<<(tot + 255) / 256, 256, 0, stream>>>(h3, l4w, l4b,
                                                           (float*)d_out, L);
    }
}
